// GPSLayer_61830349193924
// MI455X (gfx1250) — compile-verified, matrix operand order checked by reading, not run on hardware
//
#include <hip/hip_runtime.h>
#include <math.h>

// ---------------------------------------------------------------------------
// GPS layer for gfx1250 (CDNA5, wave32, WMMA bf16 16x16x32 w/ f32 accumulate)
// ---------------------------------------------------------------------------

typedef __attribute__((ext_vector_type(16))) __bf16 v16bf;
typedef __attribute__((ext_vector_type(8)))  float  v8f;

#define GN 16384   // total nodes
#define GD 128     // hidden dim
#define GE 524288  // edges
#define GB 16      // graphs
#define GH 8       // heads
#define GNP 1024   // nodes per graph
#define GDFF 512   // ffn hidden

// native bf16 convert (backend emits v_cvt; RNE)
__device__ __forceinline__ unsigned short f2bf(float f) {
  union { __bf16 b; unsigned short u; } c;
  c.b = (__bf16)f;
  return c.u;
}
__device__ __forceinline__ unsigned int pack_bf2(float lo, float hi) {
  return (unsigned int)f2bf(lo) | ((unsigned int)f2bf(hi) << 16);
}

union BF16Frag {
  v16bf v;
  unsigned short s[16];
  unsigned int u[8];
};

// K offset (within one 32-wide K step) for A/B fragment dword v.
// Wave32 layout: lanes 0-15 hold K {0..7,16..23}, lanes 16-31 hold K {8..15,24..31}.
__device__ __forceinline__ int frag_k2(int v, int hi) {
  return ((v < 4) ? 0 : 16) + hi * 8 + (v & 3) * 2;   // even; +1 is the pair
}

// ---------------------------------------------------------------------------
// Kernel 1: fused GINEConv edge stage:
//   e = edge_attr @ edge_w^T + edge_b ; msg = relu(h[src] + e) ; agg[dst] += msg
// One wave per 16-edge tile; edge_w (128x128) staged as bf16 in LDS (32 KB).
// ---------------------------------------------------------------------------
__global__ void __launch_bounds__(256)
edge_gine_kernel(const float* __restrict__ edge_attr,
                 const float* __restrict__ h,
                 const float* __restrict__ Wf,
                 const float* __restrict__ bf,
                 const int* __restrict__ src,
                 const int* __restrict__ dst,
                 float* __restrict__ agg) {
  __shared__ unsigned int Wl[GD * GD / 2];     // bf16 pairs
  __shared__ float bl[GD];
  const int tid = threadIdx.x;
  for (int i = tid; i < GD * GD / 2; i += blockDim.x)
    Wl[i] = pack_bf2(Wf[2 * i], Wf[2 * i + 1]);
  for (int i = tid; i < GD; i += blockDim.x) bl[i] = bf[i];
  __syncthreads();

  const int lane = tid & 31, wid = tid >> 5;
  const int m = lane & 15, hi = lane >> 4;
  const long e0 = ((long)blockIdx.x * 8 + wid) * 16;

  // A fragments: this wave's 16 edge rows, all 4 K-steps (K = 128)
  BF16Frag a[4];
  const float* arow = edge_attr + (e0 + m) * GD;
#pragma unroll
  for (int s = 0; s < 4; ++s)
#pragma unroll
    for (int v = 0; v < 8; ++v) {
      int k = 32 * s + frag_k2(v, hi);
      a[s].u[v] = pack_bf2(arow[k], arow[k + 1]);
    }
  int esrc[8], edst[8];
#pragma unroll
  for (int r = 0; r < 8; ++r) {
    long eg = e0 + r + 8 * hi;
    esrc[r] = src[eg];
    edst[r] = dst[eg];
  }

  for (int j = 0; j < GD / 16; ++j) {              // 8 output-channel tiles
    v8f acc = {};
    const int n = j * 16 + m;                      // this lane's output column
#pragma unroll
    for (int s = 0; s < 4; ++s) {
      BF16Frag b;
#pragma unroll
      for (int v = 0; v < 8; ++v) {
        int k = 32 * s + frag_k2(v, hi);
        b.u[v] = Wl[(n * GD + k) >> 1];
      }
      acc = __builtin_amdgcn_wmma_f32_16x16x32_bf16(
          false, a[s].v, false, b.v, (short)0, acc, false, false);
    }
    const float bias = bl[n];
    // batch the 8 h[src] gathers so the loads pipeline before the atomics
    float hv[8];
#pragma unroll
    for (int r = 0; r < 8; ++r) hv[r] = h[(long)esrc[r] * GD + n];
#pragma unroll
    for (int r = 0; r < 8; ++r) {
      float msg = hv[r] + acc[r] + bias;
      msg = msg > 0.f ? msg : 0.f;
      atomicAdd(&agg[(long)edst[r] * GD + n], msg);
    }
  }
}

// ---------------------------------------------------------------------------
// Kernel 2: generic bf16-WMMA GEMM  out[oRow] = act(A[sRow] (+addA) @ W^T + b) (+ res[oRow])
// W is [Nout][K] row-major f32 (torch Linear layout), staged as bf16 in dynamic LDS.
// Each wave owns one 16-row tile; its A tile is staged bf16 in LDS too.
// ---------------------------------------------------------------------------
__global__ void __launch_bounds__(256)
gemm_bf16_kernel(const float* __restrict__ A,
                 const float* __restrict__ addA,
                 const int* __restrict__ rowIdx,
                 const float* __restrict__ Wf,
                 const float* __restrict__ bf,
                 const float* __restrict__ res,
                 const int* __restrict__ outIdx,
                 float* __restrict__ out,
                 int K, int Nout, int act) {
  extern __shared__ char smem[];
  float* bl = (float*)smem;
  unsigned int* Wl = (unsigned int*)(smem + (size_t)Nout * 4);        // bf16 pairs
  unsigned int* At = Wl + ((size_t)Nout * K >> 1);

  const int tid = threadIdx.x;
  const int WK2 = (Nout * K) >> 1;
  for (int i = tid; i < WK2; i += blockDim.x)
    Wl[i] = pack_bf2(Wf[2 * i], Wf[2 * i + 1]);
  for (int i = tid; i < Nout; i += blockDim.x) bl[i] = bf[i];
  __syncthreads();

  const int lane = tid & 31, wid = tid >> 5;
  const int m = lane & 15, hi = lane >> 4;
  const int tile = blockIdx.x * 8 + wid;
  unsigned int* Aw = At + ((size_t)wid * 16 * K >> 1);

  // stage this wave's 16 x K A-tile as bf16 pairs (each half-wave: half the cols)
  {
    const int row = tile * 16 + m;
    const int sr = rowIdx ? rowIdx[row] : row;
    const float* ap = A + (long)sr * K;
    const float* xp = addA ? addA + (long)sr * K : nullptr;
    const int c0 = hi * (K >> 1), c1 = c0 + (K >> 1);
    for (int c = c0; c < c1; c += 2) {
      float v0 = ap[c], v1 = ap[c + 1];
      if (xp) { v0 += xp[c]; v1 += xp[c + 1]; }
      Aw[(m * K + c) >> 1] = pack_bf2(v0, v1);
    }
  }

  const int KS = K >> 5;
  for (int j = 0; j < (Nout >> 4); ++j) {
    v8f acc = {};
    const int n = j * 16 + m;
    for (int s = 0; s < KS; ++s) {
      BF16Frag af, bfg;
#pragma unroll
      for (int v = 0; v < 8; ++v) {
        int k = 32 * s + frag_k2(v, hi);
        af.u[v]  = Aw[(m * K + k) >> 1];
        bfg.u[v] = Wl[(n * K + k) >> 1];
      }
      acc = __builtin_amdgcn_wmma_f32_16x16x32_bf16(
          false, af.v, false, bfg.v, (short)0, acc, false, false);
    }
    const float bias = bl[n];
#pragma unroll
    for (int r = 0; r < 8; ++r) {
      int row = tile * 16 + r + 8 * hi;
      int orow = outIdx ? outIdx[row] : row;
      float v = acc[r] + bias;
      if (act == 1)      v = v > 0.f ? v : 0.f;                         // relu
      else if (act == 2) v = 0.5f * v * (1.0f + erff(v * 0.70710678f)); // exact gelu
      if (res) v += res[(long)orow * Nout + n];
      out[(long)orow * Nout + n] = v;
    }
  }
}

// ---------------------------------------------------------------------------
// Kernel 3: flash-style MHA, one wave per (graph, head, 16-query tile).
// head_dim = 16 (padded to K=32 for score WMMA); 32 keys per P.V WMMA.
// Row softmax stats via shfl_xor across the 16-lane half (C-layout rows).
// ---------------------------------------------------------------------------
__global__ void __launch_bounds__(32)
attn_kernel(const float* __restrict__ qkv,   // [B*NP][3*D] sorted-dense order
            float* __restrict__ ao) {        // [B*NP][D]
  __shared__ unsigned short Pt[16 * 32];
  const int lane = threadIdx.x & 31;
  const int m = lane & 15, hi = lane >> 4;
  const int bid = blockIdx.x;            // b*H*64 + h*64 + qt
  const int qt = bid & 63;
  const int hh = (bid >> 6) & 7;
  const int b  = bid >> 9;
  const long base = (long)b * GNP;
  const int q0 = qt * 16;
  const float scale = 0.25f;             // 1/sqrt(16)

  BF16Frag qa;                           // Q tile 16x32 (upper 16 K zero-pad)
  {
    const float* qrow = qkv + (base + q0 + m) * (3 * GD) + hh * 16;
#pragma unroll
    for (int v = 0; v < 8; ++v) {
      int k = frag_k2(v, hi);
      qa.u[v] = (k < 16) ? pack_bf2(qrow[k] * scale, qrow[k + 1] * scale) : 0u;
    }
  }

  float rm[8], rs[8];
#pragma unroll
  for (int r = 0; r < 8; ++r) { rm[r] = -1e30f; rs[r] = 0.f; }
  v8f oacc = {};

  for (int kt = 0; kt < GNP / 32; ++kt) {          // 32 keys per iteration
    const int k0 = kt * 32;
    v8f sacc[2];
#pragma unroll
    for (int half = 0; half < 2; ++half) {         // two 16-key score tiles
      BF16Frag kb;
      const float* krow = qkv + (base + k0 + half * 16 + m) * (3 * GD) + GD + hh * 16;
#pragma unroll
      for (int v = 0; v < 8; ++v) {
        int k = frag_k2(v, hi);
        kb.u[v] = (k < 16) ? pack_bf2(krow[k], krow[k + 1]) : 0u;
      }
      v8f z = {};
      sacc[half] = __builtin_amdgcn_wmma_f32_16x16x32_bf16(
          false, qa.v, false, kb.v, (short)0, z, false, false);
    }
    // online softmax update
    float tmax[8];
#pragma unroll
    for (int r = 0; r < 8; ++r)
      tmax[r] = fmaxf(sacc[0][r], sacc[1][r]);
    for (int d = 1; d < 16; d <<= 1)
#pragma unroll
      for (int r = 0; r < 8; ++r)
        tmax[r] = fmaxf(tmax[r], __shfl_xor(tmax[r], d, 32));
    float p0[8], p1[8], psum[8];
#pragma unroll
    for (int r = 0; r < 8; ++r) {
      float nm = fmaxf(rm[r], tmax[r]);
      float corr = __expf(rm[r] - nm);
      rm[r] = nm;
      p0[r] = __expf(sacc[0][r] - nm);
      p1[r] = __expf(sacc[1][r] - nm);
      psum[r] = p0[r] + p1[r];
      rs[r] *= corr;
      oacc[r] *= corr;
    }
    for (int d = 1; d < 16; d <<= 1)
#pragma unroll
      for (int r = 0; r < 8; ++r)
        psum[r] += __shfl_xor(psum[r], d, 32);
#pragma unroll
    for (int r = 0; r < 8; ++r) rs[r] += psum[r];

    // P (C layout) -> LDS -> A layout for the P.V product
#pragma unroll
    for (int r = 0; r < 8; ++r) {
      Pt[(r + 8 * hi) * 32 + m]      = f2bf(p0[r]);
      Pt[(r + 8 * hi) * 32 + 16 + m] = f2bf(p1[r]);
    }
    __syncthreads();  // single-wave WG: barrier is trivial; DS in-order per wave
    BF16Frag pa, vb;
#pragma unroll
    for (int v = 0; v < 8; ++v) {
      int k = frag_k2(v, hi);
      pa.u[v] = *(const unsigned int*)&Pt[m * 32 + k];
      vb.u[v] = pack_bf2(qkv[(base + k0 + k)     * (3 * GD) + 2 * GD + hh * 16 + m],
                         qkv[(base + k0 + k + 1) * (3 * GD) + 2 * GD + hh * 16 + m]);
    }
    oacc = __builtin_amdgcn_wmma_f32_16x16x32_bf16(
        false, pa.v, false, vb.v, (short)0, oacc, false, false);
    __syncthreads();
  }
#pragma unroll
  for (int r = 0; r < 8; ++r)
    ao[(base + q0 + r + 8 * hi) * GD + hh * 16 + m] = oacc[r] / rs[r];
}

// ---------------------------------------------------------------------------
// Kernel 4: LayerNorm over D=128, one wave per row.
// ---------------------------------------------------------------------------
__global__ void __launch_bounds__(256)
ln_kernel(const float* __restrict__ x, const float* __restrict__ w,
          const float* __restrict__ b, float* __restrict__ out) {
  const int lane = threadIdx.x & 31, wid = threadIdx.x >> 5;
  const long row = (long)blockIdx.x * 8 + wid;
  const float* xr = x + row * GD;
  float v[4], s = 0.f;
#pragma unroll
  for (int i = 0; i < 4; ++i) { v[i] = xr[lane + 32 * i]; s += v[i]; }
  for (int d = 1; d < 32; d <<= 1) s += __shfl_xor(s, d, 32);
  const float mu = s * (1.f / GD);
  float var = 0.f;
#pragma unroll
  for (int i = 0; i < 4; ++i) { float t = v[i] - mu; var += t * t; }
  for (int d = 1; d < 32; d <<= 1) var += __shfl_xor(var, d, 32);
  const float inv = rsqrtf(var * (1.f / GD) + 1e-5f);
#pragma unroll
  for (int i = 0; i < 4; ++i) {
    int c = lane + 32 * i;
    out[row * GD + c] = (v[i] - mu) * inv * w[c] + b[c];
  }
}

// ---------------------------------------------------------------------------
extern "C" void kernel_launch(void* const* d_in, const int* in_sizes, int n_in,
                              void* d_out, int out_size, void* d_ws, size_t ws_size,
                              hipStream_t stream) {
  const float* h        = (const float*)d_in[0];
  const float* edge_at  = (const float*)d_in[1];
  const float* edge_w   = (const float*)d_in[2];
  const float* edge_b   = (const float*)d_in[3];
  const float* gin_w1   = (const float*)d_in[4];
  const float* gin_b1   = (const float*)d_in[5];
  const float* gin_w2   = (const float*)d_in[6];
  const float* gin_b2   = (const float*)d_in[7];
  const float* in_w     = (const float*)d_in[8];
  const float* in_b     = (const float*)d_in[9];
  const float* out_w    = (const float*)d_in[10];
  const float* out_b    = (const float*)d_in[11];
  const float* ffn_w1   = (const float*)d_in[12];
  const float* ffn_b1   = (const float*)d_in[13];
  const float* ffn_w2   = (const float*)d_in[14];
  const float* ffn_b2   = (const float*)d_in[15];
  const float* ln1_w    = (const float*)d_in[16];
  const float* ln1_b    = (const float*)d_in[17];
  const float* ln2_w    = (const float*)d_in[18];
  const float* ln2_b    = (const float*)d_in[19];
  const float* ln3_w    = (const float*)d_in[20];
  const float* ln3_b    = (const float*)d_in[21];
  const int*   eidx     = (const int*)d_in[22];
  const int*   sort_idx = (const int*)d_in[24];
  float* outp = (float*)d_out;

  // workspace layout (floats); buffers aliased once their producer is consumed
  float* ws = (float*)d_ws;
  size_t o = 0;
  float* bufA = ws + o; o += (size_t)GN * GD;   // agg, then t1
  float* bufB = ws + o; o += (size_t)GN * GD;   // gin mid, then t2
  float* h1   = ws + o; o += (size_t)GN * GD;
  float* qkvb = ws + o; o += (size_t)GN * 3 * GD;
  float* aob  = ws + o; o += (size_t)GN * GD;   // attn out, then t3
  float* h2   = ws + o; o += (size_t)GN * GD;
  float* fmid = ws + o; o += (size_t)GN * GDFF;
  (void)ws_size; (void)o; (void)n_in; (void)in_sizes; (void)out_size;

  const int* src = eidx;
  const int* dst = eidx + GE;

  // allow up to ~260 KB dynamic LDS for the big GEMM stages
  hipFuncSetAttribute((const void*)gemm_bf16_kernel,
                      hipFuncAttributeMaxDynamicSharedMemorySize, 270336);

  auto gemm = [&](const float* A, const float* addA, const int* rowIdx,
                  const float* W, const float* bias, const float* res,
                  const int* outIdx, float* out, int K, int Nout, int act) {
    size_t sm = (size_t)Nout * 4 + (size_t)Nout * K * 2 + (size_t)8 * 16 * K * 2;
    gemm_bf16_kernel<<<GN / 128, 256, sm, stream>>>(A, addA, rowIdx, W, bias,
                                                    res, outIdx, out, K, Nout, act);
  };

  // 1) agg = 0 ; fused edge GEMM + gather + relu + scatter-add
  hipMemsetAsync(bufA, 0, (size_t)GN * GD * 4, stream);
  edge_gine_kernel<<<GE / 128, 256, 0, stream>>>(edge_at, h, edge_w, edge_b,
                                                 src, dst, bufA);
  // 2) GIN MLP: mid = relu((h+agg) W1^T + b1) ; t1 = h + (mid W2^T + b2)
  gemm(h, bufA, nullptr, gin_w1, gin_b1, nullptr, nullptr, bufB, GD, GD, 1);
  gemm(bufB, nullptr, nullptr, gin_w2, gin_b2, h, nullptr, bufA, GD, GD, 0);
  // 3) h1 = LN1(t1)
  ln_kernel<<<GN / 8, 256, 0, stream>>>(bufA, ln1_w, ln1_b, h1);
  // 4) qkv = h[sort_idx] @ in_proj^T + b   (sorted-dense order; graphs are full)
  gemm(h, nullptr, sort_idx, in_w, in_b, nullptr, nullptr, qkvb, GD, 3 * GD, 0);
  // 5) attention (flash, per graph/head/q-tile)
  attn_kernel<<<GB * GH * (GNP / 16), 32, 0, stream>>>(qkvb, aob);
  // 6) t2[sort_idx[i]] = h1[sort_idx[i]] + ao[i] @ out_w^T + out_b  (unsort via scatter)
  gemm(aob, nullptr, nullptr, out_w, out_b, h1, sort_idx, bufB, GD, GD, 0);
  // 7) h2 = LN2(t2)
  ln_kernel<<<GN / 8, 256, 0, stream>>>(bufB, ln2_w, ln2_b, h2);
  // 8) FFN: fmid = gelu(h2 W1^T + b1) ; t3 = h2 + fmid W2^T + b2
  gemm(h2, nullptr, nullptr, ffn_w1, ffn_b1, nullptr, nullptr, fmid, GD, GDFF, 2);
  gemm(fmid, nullptr, nullptr, ffn_w2, ffn_b2, h2, nullptr, aob, GDFF, GD, 0);
  // 9) out = LN3(t3)
  ln_kernel<<<GN / 8, 256, 0, stream>>>(aob, ln3_w, ln3_b, outp);
}
